// SimpleModel_23725399343266
// MI455X (gfx1250) — matrix operand order repaired and checked by reading, not compiled, not run on hardware
//
#include <hip/hip_runtime.h>
#include <hip/hip_bf16.h>
#include <stdint.h>

// Problem constants (match reference)
#define B_ 128
#define T_ 256
#define D_ 1280
#define C_ 100
#define KT_ 40        // D / 32 (K-iterations per GEMM)
#define WM_ 16        // batch-tile rows per workgroup
#define WN_ 64        // h-column tile per workgroup
#define NWG_M_ 8      // B / WM
#define NWG_N_ 20     // D / WN  (barrier group size)
#define GRU_THREADS 128   // 4 wave32 waves; each wave owns 16 output columns

typedef __attribute__((ext_vector_type(16))) _Float16 v16h;
typedef __attribute__((ext_vector_type(8)))  float    v8f;

union Frag16 { v16h v; unsigned u[8]; uint4 q[2]; };

__device__ __forceinline__ unsigned pk2h(float a, float b) {
  union { _Float16 h[2]; unsigned u; } x;
  x.h[0] = (_Float16)a; x.h[1] = (_Float16)b;
  return x.u;
}

// ---------------------------------------------------------------------------
// Init: zero the barrier region (first 8KB of ws) and the f16 hidden state.
// Must run every launch (ws is poisoned once and never re-initialized).
// ---------------------------------------------------------------------------
__global__ void gru_init_state(unsigned* __restrict__ bar,
                               unsigned* __restrict__ h16u) {
  int i = blockIdx.x * blockDim.x + threadIdx.x;
  if (i < 2048) bar[i] = 0u;                      // 8KB barrier/flags region
  for (int j = i; j < (B_ * D_) / 2; j += gridDim.x * blockDim.x)
    h16u[j] = 0u;                                  // h0 = 0 (f16 pairs)
}

// ---------------------------------------------------------------------------
// Pack W_ih / W_hh (f32 [3D x D], row = gate-output col) into the exact
// WMMA B-operand layout (ISA 7.12.2, 16-bit B 32x16):
//   VGPR v, lanes 0-15:  K = 2v,2v+1 (v<4), 16+2(v-4),+1 (v>=4),  N = lane
//   VGPR v, lanes 16-31: K = 8+2v,+1 (v<4), 24+2(v-4),+1 (v>=4), N = lane-16
// Packed element dword index: ((nt*KT + kt)*32 + lane)*8 + v
// so a B fragment is 32 contiguous bytes per lane -> two global_load_b128.
// ---------------------------------------------------------------------------
__global__ void gru_pack_weights(const float* __restrict__ Wih,
                                 const float* __restrict__ Whh,
                                 unsigned* __restrict__ wih_pk,
                                 unsigned* __restrict__ whh_pk) {
  const int total = (3 * D_ / 16) * KT_ * 32 * 8;  // 2,457,600 dwords/matrix
  for (int idx = blockIdx.x * blockDim.x + threadIdx.x; idx < 2 * total;
       idx += gridDim.x * blockDim.x) {
    int p = idx;
    const float* W;
    unsigned* out;
    if (p < total) { W = Wih; out = wih_pk; }
    else           { W = Whh; out = whh_pk; p -= total; }
    const int v    = p & 7;
    const int lane = (p >> 3) & 31;
    const int kn   = p >> 8;
    const int kt   = kn % KT_;
    const int nt   = kn / KT_;
    const int k    = kt * 32 + ((lane >> 4) << 3) + ((v >> 2) << 4) + 2 * (v & 3);
    const int g    = nt * 16 + (lane & 15);
    out[p] = pk2h(W[(size_t)g * D_ + k], W[(size_t)g * D_ + k + 1]);
  }
}

// ---------------------------------------------------------------------------
// Persistent GRU. Grid = 8 batch-groups x 20 column-tiles. Each step:
//   * async-stage h tile (f16, 40KB) global->LDS (ASYNCcnt path)
//   * stage x_t tile with f32->f16 conversion into LDS (overlapped)
//   * two [16x1280]x[1280x192] WMMA GEMMs (input + hidden projection)
//   * gate math in registers (rcp-based sigmoid/tanh), publish h (f16)
//   * sense-reversing barrier among the 20 WGs of the same batch group
// ---------------------------------------------------------------------------
__global__ __launch_bounds__(GRU_THREADS) void gru_persistent(
    const float* __restrict__ seq, const float* __restrict__ b_ih,
    const float* __restrict__ b_hh, const unsigned* __restrict__ wih_pk,
    const unsigned* __restrict__ whh_pk, _Float16* __restrict__ h16,
    float* __restrict__ hf, unsigned* __restrict__ bar) {
  __shared__ unsigned ldsX[WM_ * D_ / 2];  // 40 KB: x_t tile as packed f16
  __shared__ unsigned ldsH[WM_ * D_ / 2];  // 40 KB: h tile (async-staged f16)

  const int wg_m = blockIdx.x & 7;    // batch group
  const int wg_n = blockIdx.x >> 3;   // column tile (0..19)
  const int wave = threadIdx.x >> 5;
  const int lane = threadIdx.x & 31;
  const int rb   = wg_m * WM_;
  const int col  = wg_n * WN_ + wave * 16 + (lane & 15);  // this lane's N col

  unsigned* gbar = bar + wg_m * 64;   // per-group barrier, 256B apart

  // Biases for this lane's column (same for all 8 C-rows held by the lane)
  const float bias_r  = b_ih[col] + b_hh[col];
  const float bias_z  = b_ih[D_ + col] + b_hh[D_ + col];
  const float bias_in = b_ih[2 * D_ + col];
  const float bias_hn = b_hh[2 * D_ + col];

  // A-operand layout (16-bit A 16x32): lane -> row M = lane&15; the lane's
  // data is two contiguous 16B chunks: K=[klA,klA+8) and K=[klA+16,klA+24).
  const int mA  = lane & 15;
  const int klA = (lane >> 4) ? 8 : 0;
  // C/D layout: VGPR v holds row M = v (lanes 0-15) or v+8 (lanes 16-31).
  const int mC  = (lane >> 4) << 3;

  float hreg[8];                       // f32 master copy of own h slice
  #pragma unroll
  for (int v = 0; v < 8; ++v) hreg[v] = 0.f;

  const unsigned ldsh_base = (unsigned)(uintptr_t)&ldsH[0];
  const unsigned long long h16_gbase =
      (unsigned long long)(uintptr_t)(h16 + (size_t)rb * D_);
  const char* ldsXb = (const char*)ldsX;
  const char* ldsHb = (const char*)ldsH;

  for (int t = 0; t < T_; ++t) {
    // ---- async-stage h tile (WM x D f16 = 2560 x 16B) into LDS ----
    #pragma unroll 1
    for (int i = threadIdx.x; i < WM_ * D_ * 2 / 16; i += GRU_THREADS) {
      const unsigned lds_off = ldsh_base + (unsigned)i * 16u;
      const unsigned long long ga = h16_gbase + (unsigned long long)i * 16ull;
      asm volatile("global_load_async_to_lds_b128 %0, %1, off"
                   :: "v"(lds_off), "v"(ga) : "memory");
    }

    // ---- stage x_t tile (WM x D) with f32->f16 conversion (overlapped) ----
    #pragma unroll 1
    for (int i = threadIdx.x; i < WM_ * D_ / 4; i += GRU_THREADS) {
      const int e = i * 4;
      const int m = e / D_;
      const int k = e - m * D_;
      const float4 f =
          *(const float4*)(seq + ((size_t)(rb + m) * T_ + t) * D_ + k);
      *(uint2*)&ldsX[(m * D_ + k) >> 1] =
          make_uint2(pk2h(f.x, f.y), pk2h(f.z, f.w));
    }
    asm volatile("s_wait_asynccnt 0x0" ::: "memory");
    __syncthreads();

    // ---- fused GEMMs: acc[0..2] = x_t @ Wih.T (r,z,n),
    //                   acc[3..5] = h   @ Whh.T (r,z,n) ----
    v8f acc[6] = {};
    #pragma unroll 1
    for (int kt = 0; kt < KT_; ++kt) {
      Frag16 ax, ah;
      const int rowb  = mA * (D_ * 2);          // row byte offset in LDS tile
      const int kbyte = (kt * 32 + klA) * 2;    // K byte offset
      ax.q[0] = *(const uint4*)(ldsXb + rowb + kbyte);        // ds_load_b128
      ax.q[1] = *(const uint4*)(ldsXb + rowb + kbyte + 32);
      ah.q[0] = *(const uint4*)(ldsHb + rowb + kbyte);
      ah.q[1] = *(const uint4*)(ldsHb + rowb + kbyte + 32);
      #pragma unroll
      for (int g = 0; g < 3; ++g) {
        const int nt = g * 80 + wg_n * 4 + wave;  // gate-column tile index
        const size_t boff = (((size_t)nt * KT_ + kt) * 32 + lane) * 8;
        Frag16 bi, bh;
        const uint4* bpi = (const uint4*)(wih_pk + boff);
        const uint4* bph = (const uint4*)(whh_pk + boff);
        bi.q[0] = bpi[0]; bi.q[1] = bpi[1];       // 2x global_load_b128
        bh.q[0] = bph[0]; bh.q[1] = bph[1];
        acc[g] = __builtin_amdgcn_wmma_f32_16x16x32_f16(
            false, ax.v, false, bi.v, (short)0, acc[g], false, false);
        acc[3 + g] = __builtin_amdgcn_wmma_f32_16x16x32_f16(
            false, ah.v, false, bh.v, (short)0, acc[3 + g], false, false);
      }
    }

    // ---- gate math (wave-local; r,z,n columns align with this wave).
    //      rcp-based sigmoid/tanh keeps the serial epilogue short. ----
    #pragma unroll
    for (int v = 0; v < 8; ++v) {
      const float r =
          __builtin_amdgcn_rcpf(1.f + __expf(-(acc[0][v] + acc[3][v] + bias_r)));
      const float z =
          __builtin_amdgcn_rcpf(1.f + __expf(-(acc[1][v] + acc[4][v] + bias_z)));
      const float nl = acc[2][v] + bias_in + r * (acc[5][v] + bias_hn);
      const float e2 = __expf(-2.f * nl);
      const float na = (1.f - e2) * __builtin_amdgcn_rcpf(1.f + e2);  // tanh
      const float hn = (1.f - z) * na + z * hreg[v];
      hreg[v] = hn;
      h16[(size_t)(rb + v + mC) * D_ + col] = (_Float16)hn;  // publish f16 h
    }

    // ---- barrier among the 20 WGs of this batch group ----
    __syncthreads();
    if (threadIdx.x == 0) {
      __builtin_amdgcn_fence(__ATOMIC_RELEASE, "agent");
      unsigned gen = __hip_atomic_load(&gbar[1], __ATOMIC_RELAXED,
                                       __HIP_MEMORY_SCOPE_AGENT);
      unsigned arr = __hip_atomic_fetch_add(&gbar[0], 1u, __ATOMIC_ACQ_REL,
                                            __HIP_MEMORY_SCOPE_AGENT);
      if (arr == NWG_N_ - 1) {
        __hip_atomic_store(&gbar[0], 0u, __ATOMIC_RELAXED,
                           __HIP_MEMORY_SCOPE_AGENT);
        __hip_atomic_store(&gbar[1], gen + 1u, __ATOMIC_RELEASE,
                           __HIP_MEMORY_SCOPE_AGENT);
      } else {
        while (__hip_atomic_load(&gbar[1], __ATOMIC_ACQUIRE,
                                 __HIP_MEMORY_SCOPE_AGENT) == gen) {
          __builtin_amdgcn_s_sleep(2);
        }
      }
      __builtin_amdgcn_fence(__ATOMIC_ACQUIRE, "agent");
    }
    __syncthreads();
  }

  // final hidden in f32 for the scoring kernel
  #pragma unroll
  for (int v = 0; v < 8; ++v)
    hf[(size_t)(rb + v + mC) * D_ + col] = hreg[v];
}

// ---------------------------------------------------------------------------
// scores[b,c] = hidden[b,:] . candidates[b,c,:]   (one wave per (b,c) pair)
// Memory-bound: 66 MB candidate read ~= 3 us at HBM peak.
// ---------------------------------------------------------------------------
__global__ void gru_score_kernel(const float* __restrict__ hf,
                                 const float* __restrict__ cand,
                                 float* __restrict__ out) {
  const int wave = threadIdx.x >> 5, lane = threadIdx.x & 31;
  const int pair = blockIdx.x * 8 + wave;
  if (pair >= B_ * C_) return;
  const int b = pair / C_, c = pair % C_;
  const float* hp = hf + (size_t)b * D_;
  const float* cp = cand + ((size_t)b * C_ + c) * D_;
  float s = 0.f;
  #pragma unroll 4
  for (int d = lane; d < D_; d += 32) s += hp[d] * cp[d];
  #pragma unroll
  for (int off = 16; off; off >>= 1) s += __shfl_down(s, off);
  if (lane == 0) out[pair] = s;
}

// ---------------------------------------------------------------------------
extern "C" void kernel_launch(void* const* d_in, const int* in_sizes, int n_in,
                              void* d_out, int out_size, void* d_ws,
                              size_t ws_size, hipStream_t stream) {
  const float* seq  = (const float*)d_in[0];
  const float* cand = (const float*)d_in[1];
  const float* Wih  = (const float*)d_in[2];
  const float* Whh  = (const float*)d_in[3];
  const float* bih  = (const float*)d_in[4];
  const float* bhh  = (const float*)d_in[5];
  float* out = (float*)d_out;

  // Workspace layout (~21.4 MB total):
  char* ws = (char*)d_ws;
  unsigned* bar    = (unsigned*)(ws);                          // 8 KB barriers
  _Float16* h16    = (_Float16*)(ws + 8192);                   // 320 KB
  float*    hf     = (float*)(ws + 8192 + (size_t)B_ * D_ * 2);// 640 KB
  unsigned* wih_pk = (unsigned*)(ws + ((size_t)1 << 20));      // 9.4 MB
  unsigned* whh_pk = (unsigned*)(ws + ((size_t)11 << 20));     // 9.4 MB

  gru_init_state<<<320, 256, 0, stream>>>(bar, (unsigned*)h16);
  gru_pack_weights<<<4800, 256, 0, stream>>>(Wih, Whh, wih_pk, whh_pk);
  gru_persistent<<<NWG_M_ * NWG_N_, GRU_THREADS, 0, stream>>>(
      seq, bih, bhh, wih_pk, whh_pk, h16, hf, bar);
  gru_score_kernel<<<(B_ * C_ + 7) / 8, 256, 0, stream>>>(hf, cand, out);
}
